// FeedForwardDualQAttention_33621003993530
// MI455X (gfx1250) — matrix operand order repaired and checked by reading, not compiled, not run on hardware
//
#include <hip/hip_runtime.h>
#include <hip/hip_bf16.h>
#include <math.h>

// Problem constants (B=2, S=2048, D=1024, H=16, M=4, HD=64)
#define BB 2
#define SS 2048
#define DD 1024
#define HH 16
#define MM 4
#define HD 64
#define BH (BB*HH)   // 32

typedef __bf16 bf16;
typedef __attribute__((ext_vector_type(16))) __bf16 bf16x16;
typedef __attribute__((ext_vector_type(8)))  __bf16 bf16x8;
typedef __attribute__((ext_vector_type(8)))  float  floatx8;
typedef __attribute__((ext_vector_type(4)))  float  floatx4;

union BF16Frag { bf16x16 v; bf16x8 h[2]; };

static __device__ __forceinline__ bf16 f2bf(float f) {
  unsigned u = __builtin_bit_cast(unsigned, f);
  unsigned r = u + 0x7FFFu + ((u >> 16) & 1u);     // round-to-nearest-even
  unsigned short hs = (unsigned short)(r >> 16);
  return __builtin_bit_cast(bf16, hs);
}

static __device__ __forceinline__ floatx8 zero8() {
  floatx8 z = {0.f,0.f,0.f,0.f,0.f,0.f,0.f,0.f};
  return z;
}

// ---------------------------------------------------------------------------
// Kernel 1: values split-heads copy (f32 output) + transposed bf16 copy for
// the P@V B-operand (K-major).
// ---------------------------------------------------------------------------
__global__ __launch_bounds__(256) void prep_values_kernel(
    const float* __restrict__ x,          // [B][S][D]
    float* __restrict__ vals_out,         // [BH][S][HD]  (output slice 2)
    bf16*  __restrict__ valsT)            // [BH][HD][S]
{
  const int bh = blockIdx.y;              // 0..31
  const int s0 = blockIdx.x * 128;
  const int b  = bh >> 4, h = bh & 15;
  const int tid = threadIdx.x;
  for (int i = 0; i < 32; ++i) {
    int flat = i * 256 + tid;             // 0..8191 over 128 s x 64 d
    int sl = flat >> 6, d = flat & 63;
    int s = s0 + sl;
    float f = x[((size_t)(b*SS + s))*DD + h*HD + d];
    vals_out[((size_t)bh*SS + s)*HD + d] = f;
    valsT[((size_t)bh*HD + d)*SS + s] = f2bf(f);
  }
}

// ---------------------------------------------------------------------------
// Kernel 2/4: out = act @ W^T + bias, bf16 WMMA, f32 accumulate.
//   mode 0: write bf16 q_row to ws      ([BH][S][HD])
//   mode 1: write f32  q_col to d_out   ([BH][S][HD])
//   mode 2: write f32  gelu(out) to d_out ([M][1024] row-major)
// Block = 256 threads (8 waves), tile 64x64, K-step 32.
// ---------------------------------------------------------------------------
__global__ __launch_bounds__(256) void gemm_wmma_kernel(
    const float* __restrict__ act,        // [Mrows][1024]
    const float* __restrict__ wgt,        // [1024][1024] (row n = output col)
    const float* __restrict__ bias,       // [1024]
    float* __restrict__ out_f32,
    bf16*  __restrict__ out_bf16,
    int mode)
{
  __shared__ __align__(16) bf16 Al[64][40];
  __shared__ __align__(16) bf16 Bl[64][40];

  const int tid  = threadIdx.x;
  const int wv   = tid >> 5;
  const int lane = tid & 31;
  const int l    = lane & 15;
  const int hi   = lane >> 4;

  const int m0 = blockIdx.x * 64;
  const int n0 = blockIdx.y * 64;
  const int wrow = (wv >> 1) * 16;        // 0,16,32,48
  const int wcol = (wv & 1) * 32;         // 0,32

  floatx8 acc0 = zero8();
  floatx8 acc1 = zero8();

  for (int k0 = 0; k0 < 1024; k0 += 32) {
    // cooperative f32 -> bf16 staging (64x32 each of A and B)
    {
      int rowh = tid >> 3;                // 0..31
      int quad = (tid & 7) * 4;           // 0..28
      for (int rr = 0; rr < 64; rr += 32) {
        int r = rowh + rr;
        floatx4 a4 = *(const floatx4*)(act + (size_t)(m0 + r)*1024 + k0 + quad);
        Al[r][quad+0] = f2bf(a4.x); Al[r][quad+1] = f2bf(a4.y);
        Al[r][quad+2] = f2bf(a4.z); Al[r][quad+3] = f2bf(a4.w);
        floatx4 b4 = *(const floatx4*)(wgt + (size_t)(n0 + r)*1024 + k0 + quad);
        Bl[r][quad+0] = f2bf(b4.x); Bl[r][quad+1] = f2bf(b4.y);
        Bl[r][quad+2] = f2bf(b4.z); Bl[r][quad+3] = f2bf(b4.w);
      }
    }
    __syncthreads();

    // A fragment: rows wrow..wrow+15, K 0..31 (16-bit A layout: K = hi*8 blocks)
    BF16Frag af;
    af.h[0] = *(const bf16x8*)&Al[wrow + l][hi*8];
    af.h[1] = *(const bf16x8*)&Al[wrow + l][16 + hi*8];
    // B fragments: cols wcol..+15 and wcol+16..+31 (16-bit B layout: K = hi*8 blocks)
    BF16Frag b0, b1;
    b0.h[0] = *(const bf16x8*)&Bl[wcol + l][hi*8];
    b0.h[1] = *(const bf16x8*)&Bl[wcol + l][16 + hi*8];
    b1.h[0] = *(const bf16x8*)&Bl[wcol + 16 + l][hi*8];
    b1.h[1] = *(const bf16x8*)&Bl[wcol + 16 + l][16 + hi*8];

    acc0 = __builtin_amdgcn_wmma_f32_16x16x32_bf16(false, af.v, false, b0.v,
                                                   (short)0, acc0, false, false);
    acc1 = __builtin_amdgcn_wmma_f32_16x16x32_bf16(false, af.v, false, b1.v,
                                                   (short)0, acc1, false, false);
    __syncthreads();
  }

  // Epilogue. C layout: VGPR r -> row (r + 8*hi), col = lane%16 within frag.
  const int colb = n0 + wcol + l;
  #pragma unroll
  for (int r = 0; r < 8; ++r) {
    int row = m0 + wrow + r + 8*hi;
    #pragma unroll
    for (int f = 0; f < 2; ++f) {
      int col = colb + 16*f;
      float v = (f ? acc1[r] : acc0[r]) + bias[col];
      if (mode == 0) {
        int h = col >> 6, d = col & 63;
        int b = row >> 11, s = row & 2047;
        out_bf16[(((size_t)(b*HH + h))*SS + s)*HD + d] = f2bf(v);
      } else if (mode == 1) {
        int h = col >> 6, d = col & 63;
        int b = row >> 11, s = row & 2047;
        out_f32[(((size_t)(b*HH + h))*SS + s)*HD + d] = v;
      } else {
        float g = 0.5f * v * (1.0f + erff(v * 0.7071067811865475f));
        out_f32[(size_t)row*1024 + col] = g;
      }
    }
  }
}

// ---------------------------------------------------------------------------
// Kernel 3: fused causal attention + memory slots, flash-style online softmax.
// One wave per 16-query tile; block = 128 threads (4 independent waves).
// ---------------------------------------------------------------------------
__global__ __launch_bounds__(128) void attn_kernel(
    const bf16*  __restrict__ q_row,   // [BH][S][HD]  bf16
    const float* __restrict__ q_col,   // [BH][S][HD]  f32 (output slice 1)
    const bf16*  __restrict__ valsT,   // [BH][HD][S]  bf16
    const float* __restrict__ past_q,  // [BH][S][M][HD]
    const float* __restrict__ past_v,  // [BH][S][M][HD]
    float* __restrict__ merged)        // [B][S][D]
{
  __shared__ __align__(16) bf16 Pl[4][16][40];   // per-wave P transpose scratch

  const int tid  = threadIdx.x;
  const int wv   = tid >> 5;
  const int lane = tid & 31;
  const int l    = lane & 15;
  const int hi   = lane >> 4;

  const int gw = blockIdx.x * 4 + wv;
  const int q0 = (gw & 127) * 16;
  const int bh = gw >> 7;

  const bf16* qr_bh = q_row + (size_t)bh * SS * HD;
  const bf16* vT_bh = valsT + (size_t)bh * HD * SS;

  // Q A-fragments for both K=32 halves of HD=64 (loaded once)
  BF16Frag qa[2];
  #pragma unroll
  for (int dc = 0; dc < 2; ++dc) {
    const bf16* base = qr_bh + (size_t)(q0 + l)*HD + dc*32;
    qa[dc].h[0] = *(const bf16x8*)(base + hi*8);
    qa[dc].h[1] = *(const bf16x8*)(base + 16 + hi*8);
  }

  floatx8 ctx[4] = { zero8(), zero8(), zero8(), zero8() };
  float mi[8], li[8];
  #pragma unroll
  for (int r = 0; r < 8; ++r) { mi[r] = -1e30f; li[r] = 0.f; }

  const int nkt = (q0 + 16 + 31) >> 5;          // causal key-tile bound
  for (int kt = 0; kt < nkt; ++kt) {
    const int k0 = kt * 32;
    if (kt + 1 < nkt)
      __builtin_prefetch(qr_bh + (size_t)(k0 + 32 + l)*HD, 0, 0);

    // ---- scores: S(16q x 32k) via 4 WMMAs
    floatx8 s0 = zero8(), s1 = zero8();
    #pragma unroll
    for (int dc = 0; dc < 2; ++dc) {
      BF16Frag kb0, kb1;
      const bf16* p0 = qr_bh + (size_t)(k0 + l)*HD + dc*32 + hi*8;
      kb0.h[0] = *(const bf16x8*)p0;
      kb0.h[1] = *(const bf16x8*)(p0 + 16);
      const bf16* p1 = qr_bh + (size_t)(k0 + 16 + l)*HD + dc*32 + hi*8;
      kb1.h[0] = *(const bf16x8*)p1;
      kb1.h[1] = *(const bf16x8*)(p1 + 16);
      s0 = __builtin_amdgcn_wmma_f32_16x16x32_bf16(false, qa[dc].v, false, kb0.v,
                                                   (short)0, s0, false, false);
      s1 = __builtin_amdgcn_wmma_f32_16x16x32_bf16(false, qa[dc].v, false, kb1.v,
                                                   (short)0, s1, false, false);
    }

    // ---- causal mask, scale, online softmax update
    #pragma unroll
    for (int r = 0; r < 8; ++r) {
      int row = q0 + r + 8*hi;
      int c0 = k0 + l, c1 = c0 + 16;
      float v0 = (c0 <= row) ? s0[r]*0.125f : -1e30f;
      float v1 = (c1 <= row) ? s1[r]*0.125f : -1e30f;
      float rm = fmaxf(v0, v1);
      rm = fmaxf(rm, __shfl_xor(rm, 1));
      rm = fmaxf(rm, __shfl_xor(rm, 2));
      rm = fmaxf(rm, __shfl_xor(rm, 4));
      rm = fmaxf(rm, __shfl_xor(rm, 8));
      float mn = fmaxf(mi[r], rm);
      float al = __expf(mi[r] - mn);
      float p0 = __expf(v0 - mn);
      float p1 = __expf(v1 - mn);
      float rs = p0 + p1;
      rs += __shfl_xor(rs, 1);
      rs += __shfl_xor(rs, 2);
      rs += __shfl_xor(rs, 4);
      rs += __shfl_xor(rs, 8);
      li[r] = li[r]*al + rs;
      mi[r] = mn;
      Pl[wv][r + 8*hi][l]      = f2bf(p0);
      Pl[wv][r + 8*hi][16 + l] = f2bf(p1);
      ctx[0][r] *= al; ctx[1][r] *= al; ctx[2][r] *= al; ctx[3][r] *= al;
    }
    asm volatile("s_wait_dscnt 0x0" ::: "memory");   // wave-local LDS RAW

    // ---- P (C layout) -> A layout via LDS
    BF16Frag pf;
    pf.h[0] = *(const bf16x8*)&Pl[wv][l][hi*8];
    pf.h[1] = *(const bf16x8*)&Pl[wv][l][16 + hi*8];

    // ---- ctx(16 x 64) += P(16x32) @ V(32x64): 4 WMMAs from transposed V
    #pragma unroll
    for (int dc = 0; dc < 4; ++dc) {
      BF16Frag vf;
      const bf16* vb = vT_bh + (size_t)(dc*16 + l)*SS + k0 + hi*8;
      vf.h[0] = *(const bf16x8*)vb;
      vf.h[1] = *(const bf16x8*)(vb + 16);
      ctx[dc] = __builtin_amdgcn_wmma_f32_16x16x32_bf16(false, pf.v, false, vf.v,
                                                        (short)0, ctx[dc], false, false);
    }
  }

  // ---- merge M=4 memory slots into the same online softmax, normalize, store
  const float* qc_bh = q_col  + (size_t)bh * SS * HD;
  const float* pq_bh = past_q + (size_t)bh * SS * MM * HD;
  const float* pv_bh = past_v + (size_t)bh * SS * MM * HD;
  const int b = bh >> 4, h = bh & 15;

  #pragma unroll
  for (int r = 0; r < 8; ++r) {
    int row = q0 + r + 8*hi;
    const float* qcr = qc_bh + (size_t)row * HD;
    float ms[4];
    #pragma unroll
    for (int m = 0; m < 4; ++m) {
      const float* pq = pq_bh + ((size_t)row*MM + m)*HD;
      float part = 0.f;
      #pragma unroll
      for (int t = 0; t < 4; ++t) {
        int d = l*4 + t;
        part += qcr[d] * pq[d];
      }
      part += __shfl_xor(part, 1);
      part += __shfl_xor(part, 2);
      part += __shfl_xor(part, 4);
      part += __shfl_xor(part, 8);
      ms[m] = part * 0.125f;
    }
    float mn = mi[r];
    #pragma unroll
    for (int m = 0; m < 4; ++m) mn = fmaxf(mn, ms[m]);
    float al = __expf(mi[r] - mn);
    float pm[4]; float rs = 0.f;
    #pragma unroll
    for (int m = 0; m < 4; ++m) { pm[m] = __expf(ms[m] - mn); rs += pm[m]; }
    float inv = 1.0f / (li[r]*al + rs);

    #pragma unroll
    for (int dc = 0; dc < 4; ++dc) {
      int d = dc*16 + l;
      float add = 0.f;
      #pragma unroll
      for (int m = 0; m < 4; ++m)
        add += pm[m] * pv_bh[((size_t)row*MM + m)*HD + d];
      float v = (ctx[dc][r]*al + add) * inv;
      merged[((size_t)(b*SS + row))*DD + h*HD + d] = v;
    }
  }
}

// ---------------------------------------------------------------------------
extern "C" void kernel_launch(void* const* d_in, const int* in_sizes, int n_in,
                              void* d_out, int out_size, void* d_ws, size_t ws_size,
                              hipStream_t stream) {
  (void)in_sizes; (void)n_in; (void)out_size; (void)ws_size;

  const float* x      = (const float*)d_in[0];
  const float* past_q = (const float*)d_in[1];
  const float* past_v = (const float*)d_in[2];
  const float* Wr     = (const float*)d_in[3];
  const float* br     = (const float*)d_in[4];
  const float* Wc     = (const float*)d_in[5];
  const float* bc     = (const float*)d_in[6];
  const float* Wo     = (const float*)d_in[7];
  const float* bo     = (const float*)d_in[8];

  float* out      = (float*)d_out;
  float* out_y    = out;                         // [B][S][D]
  float* out_qcol = out + (size_t)BH*SS*HD;      // [BH][S][HD]
  float* out_vals = out + 2*(size_t)BH*SS*HD;    // [BH][S][HD]

  char*  ws       = (char*)d_ws;
  bf16*  q_row_bf = (bf16*)ws;                       //  8 MB
  bf16*  valsT    = (bf16*)(ws + (size_t)(8u << 20)); //  8 MB
  float* merged   = (float*)(ws + (size_t)(16u << 20)); // 16 MB

  prep_values_kernel<<<dim3(SS/128, BH), 256, 0, stream>>>(x, out_vals, valsT);
  gemm_wmma_kernel<<<dim3((BB*SS)/64, DD/64), 256, 0, stream>>>(
      x, Wr, br, nullptr, q_row_bf, 0);
  gemm_wmma_kernel<<<dim3((BB*SS)/64, DD/64), 256, 0, stream>>>(
      x, Wc, bc, out_qcol, nullptr, 1);
  attn_kernel<<<dim3((BH*(SS/16))/4), 128, 0, stream>>>(
      q_row_bf, out_qcol, valsT, past_q, past_v, merged);
  gemm_wmma_kernel<<<dim3((BB*SS)/64, DD/64), 256, 0, stream>>>(
      merged, Wo, bo, out_y, nullptr, 2);
}
